// PathGNNEncoder_74406013436400
// MI455X (gfx1250) — matrix operand order repaired and checked by reading, not compiled, not run on hardware
//
#include <hip/hip_runtime.h>
#include <hip/hip_bf16.h>

// ---------------------------------------------------------------------------
// PathGNN encoder for MI455X (gfx1250, wave32, WMMA).
//   - dense GEMMs     : v_wmma_f32_16x16x4_f32   (full f32 precision)
//   - masks@edge_emb  : v_wmma_f32_16x16x32_bf16 (masks are 0/1 -> exact)
//   - LDS tiles       : transposed B + ld=72 padding -> ds_load_b128, no
//                       8-way bank conflicts (144B row -> bank stride 36)
//   - 410MB mask read : non-temporal 16B loads + global_prefetch_b8
//   - edge indices    : async global->LDS copy (ASYNCcnt) when available
// ---------------------------------------------------------------------------

typedef __attribute__((ext_vector_type(16))) __bf16 v16bf;
typedef __attribute__((ext_vector_type(8)))  __bf16 v8bf;
typedef __attribute__((ext_vector_type(4)))  __bf16 v4bf;
typedef __attribute__((ext_vector_type(8)))  float  v8f;
typedef __attribute__((ext_vector_type(4)))  float  fx4;
typedef __attribute__((ext_vector_type(2)))  float  v2f;
typedef __attribute__((ext_vector_type(4)))  int    v4i;

#define EMB 64
#define LDP 72  // padded LDS leading dim (bf16 elems): 144B rows, conflict-free

#if __has_builtin(__builtin_amdgcn_global_load_async_to_lds_b128) && \
    __has_builtin(__builtin_amdgcn_s_wait_asynccnt)
#define HAVE_ASYNC_LDS 1
#endif

// generic LDS pointer -> AS(3) v4i* (ISA aperture: generic LDS address low
// 32 bits are the LDS offset, so the integer round-trip is value-preserving).
__device__ __forceinline__ __attribute__((address_space(3))) v4i* lds_v4i(
    void* p) {
  return (__attribute__((address_space(3))) v4i*)(unsigned)(unsigned long long)p;
}

// ---- ordered-uint encoding so atomicMax(uint) == max(float) ----
__device__ __forceinline__ unsigned fkey(float f) {
  unsigned u = __float_as_uint(f);
  return (u >> 31) ? ~u : (u | 0x80000000u);
}
__device__ __forceinline__ float funkey(unsigned k) {
  unsigned u = (k >> 31) ? (k & 0x7FFFFFFFu) : ~k;
  return __uint_as_float(u);
}

// ---- WMMA bf16 fragment loaders (ISA 7.12.2 wave32 layouts) ----
// A 16x32 bf16: lane l -> row (l&15); elem i -> K = (i&7) + 8*(l>>4) + 16*(i>>3)
// => two contiguous 16B chunks per lane -> ds_load_b128 x2
__device__ __forceinline__ v16bf ldA_bf16(const __bf16* sm, int ld, int row0,
                                          int k0, int lane) {
  const int r = row0 + (lane & 15), hx = lane >> 4;
  const __bf16* p = sm + r * ld + k0 + 8 * hx;
  v8bf lo = *(const v8bf*)p;
  v8bf hi = *(const v8bf*)(p + 16);
  return __builtin_shufflevector(lo, hi, 0, 1, 2, 3, 4, 5, 6, 7, 8, 9, 10, 11,
                                 12, 13, 14, 15);
}
// B 32x16 bf16 from TRANSPOSED tile Bt[col][k]: lane l -> col (l&15);
// elem i -> K = k0 + (i&7) + 8*(l>>4) + 16*(i>>3)  => ds_load_b128 x2
__device__ __forceinline__ v16bf ldB_bf16T(const __bf16* smT, int ld, int k0,
                                           int col0, int lane) {
  const int c = col0 + (lane & 15), hx = lane >> 4;
  const __bf16* p = smT + c * ld + k0 + 8 * hx;
  v8bf lo = *(const v8bf*)p;
  v8bf hi = *(const v8bf*)(p + 16);
  return __builtin_shufflevector(lo, hi, 0, 1, 2, 3, 4, 5, 6, 7, 8, 9, 10, 11,
                                 12, 13, 14, 15);
}

// ---------------------------------------------------------------------------
__global__ void zero_kernel(unsigned* __restrict__ p, size_t n) {
  size_t i  = (size_t)blockIdx.x * blockDim.x + threadIdx.x;
  size_t st = (size_t)gridDim.x * blockDim.x;
  for (; i < n; i += st) p[i] = 0u;
}

// ---------------------------------------------------------------------------
// xl = X@Wl+bl, xr = X@Wr+br in one pass (shares the A operand).
// f32 WMMA 16x16x4. Block = 256 thr = 8 waves, 128 rows per block.
// M % 16 == 0 so the early-out is wave-uniform (EXEC stays all-ones).
// ---------------------------------------------------------------------------
template <int K>
__global__ void __launch_bounds__(256) dual_gemm_kernel(
    const float* __restrict__ X,
    const float* __restrict__ Wl, const float* __restrict__ bl,
    const float* __restrict__ Wr, const float* __restrict__ br,
    float* __restrict__ outL, float* __restrict__ outR, int M) {
  const int wave = threadIdx.x >> 5, lane = threadIdx.x & 31;
  const int row0 = blockIdx.x * 128 + wave * 16;
  if (row0 >= M) return;  // wave-uniform
  const int hx = lane >> 4, lc = lane & 15;
  const size_t rr = (size_t)(row0 + lc);

  v8f accL[4], accR[4];
#pragma unroll
  for (int n = 0; n < 4; ++n)
#pragma unroll
    for (int j = 0; j < 8; ++j) { accL[n][j] = 0.f; accR[n][j] = 0.f; }

  for (int k0 = 0; k0 < K; k0 += 4) {
    // A 16x4 f32: lane l -> row l&15; elem i -> K = i + 2*(l>>4)
    v2f a;
    a[0] = X[rr * K + k0 + 2 * hx];
    a[1] = X[rr * K + k0 + 2 * hx + 1];
#pragma unroll
    for (int n = 0; n < 4; ++n) {
      v2f bLf, bRf;  // B 4x16 f32: lane l -> col l&15; elem i -> K = i+2*(l>>4)
      bLf[0] = Wl[(k0 + 2 * hx) * EMB + n * 16 + lc];
      bLf[1] = Wl[(k0 + 2 * hx + 1) * EMB + n * 16 + lc];
      accL[n] = __builtin_amdgcn_wmma_f32_16x16x4_f32(
          false, a, false, bLf, (short)0, accL[n], false, false);
      bRf[0] = Wr[(k0 + 2 * hx) * EMB + n * 16 + lc];
      bRf[1] = Wr[(k0 + 2 * hx + 1) * EMB + n * 16 + lc];
      accR[n] = __builtin_amdgcn_wmma_f32_16x16x4_f32(
          false, a, false, bRf, (short)0, accR[n], false, false);
    }
  }
#pragma unroll
  for (int n = 0; n < 4; ++n)
#pragma unroll
    for (int j = 0; j < 8; ++j) {
      const int r = row0 + j + 8 * hx, c = n * 16 + lc;  // C/D layout 7.12.2
      outL[(size_t)r * EMB + c] = accL[n][j] + bl[c];
      outR[(size_t)r * EMB + c] = accR[n][j] + br[c];
    }
}

// ---------------------------------------------------------------------------
// Per-edge attention logits + running segment max (one wave per edge).
// xl/xr (12.8 MB each) are L2-resident -> gathers hit the 192MB L2.
// ---------------------------------------------------------------------------
__global__ void __launch_bounds__(256) edge_score_kernel(
    const float* __restrict__ xl, const float* __restrict__ xr,
    const int* __restrict__ ei, const float* __restrict__ att,
    float* __restrict__ escore, unsigned* __restrict__ emaxk, int E, int ESL) {
  const int gw = (int)((blockIdx.x * blockDim.x + threadIdx.x) >> 5);
  const int lane = threadIdx.x & 31;
  if (gw >= ESL) return;
  int s, d;
  if (gw < E) { s = ei[gw]; d = ei[E + gw]; } else { s = d = gw - E; }
  const float2 a = *(const float2*)(xl + (size_t)s * EMB + lane * 2);
  const float2 b = *(const float2*)(xr + (size_t)d * EMB + lane * 2);
  float m0 = a.x + b.x, m1 = a.y + b.y;
  m0 = m0 > 0.f ? m0 : 0.2f * m0;  // leaky_relu(0.2)
  m1 = m1 > 0.f ? m1 : 0.2f * m1;
  float p = m0 * att[lane * 2] + m1 * att[lane * 2 + 1];
#pragma unroll
  for (int off = 16; off > 0; off >>= 1) p += __shfl_xor(p, off, 32);
  if (lane == 0) {
    escore[gw] = p;
    atomicMax(&emaxk[d], fkey(p));
  }
}

__global__ void edge_expden_kernel(float* __restrict__ escore,
                                   const int* __restrict__ ei,
                                   const unsigned* __restrict__ emaxk,
                                   float* __restrict__ den, int E, int ESL) {
  const int i = blockIdx.x * blockDim.x + threadIdx.x;
  if (i >= ESL) return;
  const int d = (i < E) ? ei[E + i] : (i - E);
  const float ex = __expf(escore[i] - funkey(emaxk[d]));
  escore[i] = ex;
  atomicAdd(&den[d], ex);
}

__global__ void __launch_bounds__(256) edge_agg_kernel(
    const float* __restrict__ escore, const float* __restrict__ den,
    const float* __restrict__ xl, const int* __restrict__ ei,
    float* __restrict__ agg, int E, int ESL) {
  const int gw = (int)((blockIdx.x * blockDim.x + threadIdx.x) >> 5);
  const int lane = threadIdx.x & 31;
  if (gw >= ESL) return;
  int s, d;
  if (gw < E) { s = ei[gw]; d = ei[E + gw]; } else { s = d = gw - E; }
  const float alpha = escore[gw] / den[d];
  const float2 v = *(const float2*)(xl + (size_t)s * EMB + lane * 2);
  atomicAdd(&agg[(size_t)d * EMB + lane * 2],     alpha * v.x);
  atomicAdd(&agg[(size_t)d * EMB + lane * 2 + 1], alpha * v.y);
}

__global__ void bias_relu_kernel(float* __restrict__ h,
                                 const float* __restrict__ b, int total) {
  const int i = blockIdx.x * blockDim.x + threadIdx.x;
  if (i < total) h[i] = fmaxf(h[i] + b[i & (EMB - 1)], 0.f);
}

// ---------------------------------------------------------------------------
// path_emb[128,64] += masks[:,chunk] @ edge_emb[chunk,64], chunk = 64 edges.
// Streaming persistent-accumulator bf16 WMMA kernel. 410MB mask read is the
// HBM roofline term (~18us @ 23.3 TB/s): non-temporal so it cannot evict the
// L2-resident y2 that is gathered per edge.
// ---------------------------------------------------------------------------
__global__ void __launch_bounds__(256) path_accum_kernel(
    const float* __restrict__ y, const int* __restrict__ ei,
    const float* __restrict__ masks, float* __restrict__ path_emb, int E) {
  __shared__ __bf16 Asm[128 * LDP];  // [path][edge-in-chunk]      (A operand)
  __shared__ __bf16 Bt[64 * LDP];    // [dim][edge-in-chunk] transposed (B)
  __shared__ int    EiSm[128];       // src[0:64) | dst[64:128) for this chunk
  const int tid = threadIdx.x, wave = tid >> 5, lane = tid & 31;
  const int hx = lane >> 4, lc = lane & 15;

  v8f acc[4];
#pragma unroll
  for (int n = 0; n < 4; ++n)
#pragma unroll
    for (int j = 0; j < 8; ++j) acc[n][j] = 0.f;

  const int nChunks = E >> 6;  // E % 64 == 0
  for (int c = blockIdx.x; c < nChunks; c += gridDim.x) {
    const int e0 = c << 6;
    // ---- stage the 64 edge indices (src+dst) into LDS ----
#if HAVE_ASYNC_LDS
    if (wave == 0) {
      // lanes 0..15 copy src ids, lanes 16..31 copy dst ids, 16B per lane
      const int* gsrc = (lane < 16) ? (ei + e0 + lane * 4)
                                    : (ei + E + e0 + (lane - 16) * 4);
      __builtin_amdgcn_global_load_async_to_lds_b128(
          (v4i*)gsrc, lds_v4i(&EiSm[lane * 4]), 0, 0);
      __builtin_amdgcn_s_wait_asynccnt(0);
    }
#else
    if (tid < 128)
      EiSm[tid] = (tid < 64) ? ei[e0 + tid] : ei[E + e0 + tid - 64];
#endif
    __syncthreads();
    {  // stage edge_emb chunk TRANSPOSED: thread = (edge k, 16-dim slab)
      const int k = tid >> 2, d0 = (tid & 3) << 4;
      const int s = EiSm[k], d = EiSm[64 + k];
      const fx4* ys = (const fx4*)(y + (size_t)s * EMB + d0);
      const fx4* yd = (const fx4*)(y + (size_t)d * EMB + d0);
#pragma unroll
      for (int q = 0; q < 4; ++q) {
        const fx4 va = ys[q], vb = yd[q];
#pragma unroll
        for (int i = 0; i < 4; ++i)
          Bt[(d0 + q * 4 + i) * LDP + k] = (__bf16)(0.5f * (va[i] + vb[i]));
      }
    }
    {  // stage mask tile: 128 paths x 64 edges, NT 16B loads, 8B LDS stores
      const int p = tid >> 1, kb = (tid & 1) << 5;
      const float* mrow = masks + (size_t)p * E + e0 + kb;
#pragma unroll
      for (int q = 0; q < 8; ++q) {
        const fx4 m = __builtin_nontemporal_load((const fx4*)mrow + q);
        v4bf o;
#pragma unroll
        for (int i = 0; i < 4; ++i) o[i] = (__bf16)m[i];
        *(v4bf*)&Asm[p * LDP + kb + q * 4] = o;
      }
      if (c + (int)gridDim.x < nChunks)  // -> global_prefetch_b8
        __builtin_prefetch(
            masks + (size_t)p * E + (((size_t)c + gridDim.x) << 6) + kb, 0, 0);
    }
    __syncthreads();
    const int row0 = wave << 4;  // wave owns 16 path rows
#pragma unroll
    for (int ks = 0; ks < 2; ++ks) {
      const v16bf a = ldA_bf16(Asm, LDP, row0, ks * 32, lane);
#pragma unroll
      for (int n = 0; n < 4; ++n) {
        const v16bf b = ldB_bf16T(Bt, LDP, ks * 32, n * 16, lane);
        acc[n] = __builtin_amdgcn_wmma_f32_16x16x32_bf16(
            false, a, false, b, (short)0, acc[n], false, false);
      }
    }
    __syncthreads();
  }
#pragma unroll
  for (int n = 0; n < 4; ++n)
#pragma unroll
    for (int j = 0; j < 8; ++j)
      atomicAdd(&path_emb[(wave * 16 + j + 8 * hx) * EMB + n * 16 + lc],
                acc[n][j]);
}

// ---------------------------------------------------------------------------
// out = path_emb[128,64] @ Wro[64,64] + bro  (one block, f32 WMMA)
// ---------------------------------------------------------------------------
__global__ void __launch_bounds__(256) final_gemm_kernel(
    const float* __restrict__ pe, const float* __restrict__ W,
    const float* __restrict__ b, float* __restrict__ out) {
  const int wave = threadIdx.x >> 5, lane = threadIdx.x & 31;
  const int hx = lane >> 4, lc = lane & 15;
  const int row0 = wave * 16;  // 8 waves cover P=128 rows
  const int rr = row0 + lc;

  v8f acc[4];
#pragma unroll
  for (int n = 0; n < 4; ++n)
#pragma unroll
    for (int j = 0; j < 8; ++j) acc[n][j] = 0.f;

  for (int k0 = 0; k0 < EMB; k0 += 4) {
    v2f a;
    a[0] = pe[rr * EMB + k0 + 2 * hx];
    a[1] = pe[rr * EMB + k0 + 2 * hx + 1];
#pragma unroll
    for (int n = 0; n < 4; ++n) {
      v2f bw;
      bw[0] = W[(k0 + 2 * hx) * EMB + n * 16 + lc];
      bw[1] = W[(k0 + 2 * hx + 1) * EMB + n * 16 + lc];
      acc[n] = __builtin_amdgcn_wmma_f32_16x16x4_f32(
          false, a, false, bw, (short)0, acc[n], false, false);
    }
  }
#pragma unroll
  for (int n = 0; n < 4; ++n)
#pragma unroll
    for (int j = 0; j < 8; ++j)
      out[(row0 + j + 8 * hx) * EMB + n * 16 + lc] = acc[n][j] + b[n * 16 + lc];
}

// ---------------------------------------------------------------------------
extern "C" void kernel_launch(void* const* d_in, const int* in_sizes, int n_in,
                              void* d_out, int out_size, void* d_ws,
                              size_t ws_size, hipStream_t stream) {
  (void)n_in; (void)out_size; (void)ws_size;
  const float* x     = (const float*)d_in[0];
  const int*   ei    = (const int*)d_in[1];
  const float* masks = (const float*)d_in[2];
  const float* Wl1 = (const float*)d_in[3];
  const float* bl1 = (const float*)d_in[4];
  const float* Wr1 = (const float*)d_in[5];
  const float* br1 = (const float*)d_in[6];
  const float* att1 = (const float*)d_in[7];
  const float* b1  = (const float*)d_in[8];
  const float* Wl2 = (const float*)d_in[9];
  const float* bl2 = (const float*)d_in[10];
  const float* Wr2 = (const float*)d_in[11];
  const float* br2 = (const float*)d_in[12];
  const float* att2 = (const float*)d_in[13];
  const float* b2  = (const float*)d_in[14];
  const float* Wro = (const float*)d_in[15];
  const float* bro = (const float*)d_in[16];

  const int N   = in_sizes[0] / 128;  // 50000
  const int E   = in_sizes[1] / 2;    // 800000
  const int ESL = E + N;              // edges + self loops

  // workspace layout (~55 MB)
  float*    ws    = (float*)d_ws;
  float*    xl    = ws;
  float*    xr    = xl + (size_t)N * EMB;
  float*    h1    = xr + (size_t)N * EMB;
  float*    h2    = h1 + (size_t)N * EMB;
  float*    esc   = h2 + (size_t)N * EMB;
  unsigned* emaxk = (unsigned*)(esc + ESL);
  float*    den   = (float*)(emaxk + N);
  float*    pemb  = den + N;

  const dim3 blk(256);
  const dim3 zgrid(2048);
  const int gemmBlocks  = (N + 127) / 128;
  const int edgeWaveBlk = (ESL + 7) / 8;     // 8 waves (edges) per block
  const int edgeThrBlk  = (ESL + 255) / 256;
  const int nodeElemBlk = (N * EMB + 255) / 256;

  // ---------------- Layer 1 ----------------
  zero_kernel<<<zgrid, blk, 0, stream>>>((unsigned*)h1, (size_t)N * EMB);
  zero_kernel<<<zgrid, blk, 0, stream>>>(emaxk, (size_t)N);
  zero_kernel<<<zgrid, blk, 0, stream>>>((unsigned*)den, (size_t)N);
  dual_gemm_kernel<128><<<gemmBlocks, blk, 0, stream>>>(
      x, Wl1, bl1, Wr1, br1, xl, xr, N);
  edge_score_kernel<<<edgeWaveBlk, blk, 0, stream>>>(xl, xr, ei, att1, esc,
                                                     emaxk, E, ESL);
  edge_expden_kernel<<<edgeThrBlk, blk, 0, stream>>>(esc, ei, emaxk, den, E,
                                                     ESL);
  edge_agg_kernel<<<edgeWaveBlk, blk, 0, stream>>>(esc, den, xl, ei, h1, E,
                                                   ESL);
  bias_relu_kernel<<<nodeElemBlk, blk, 0, stream>>>(h1, b1, N * EMB);

  // ---------------- Layer 2 ----------------
  zero_kernel<<<zgrid, blk, 0, stream>>>((unsigned*)h2, (size_t)N * EMB);
  zero_kernel<<<zgrid, blk, 0, stream>>>(emaxk, (size_t)N);
  zero_kernel<<<zgrid, blk, 0, stream>>>((unsigned*)den, (size_t)N);
  dual_gemm_kernel<64><<<gemmBlocks, blk, 0, stream>>>(
      h1, Wl2, bl2, Wr2, br2, xl, xr, N);
  edge_score_kernel<<<edgeWaveBlk, blk, 0, stream>>>(xl, xr, ei, att2, esc,
                                                     emaxk, E, ESL);
  edge_expden_kernel<<<edgeThrBlk, blk, 0, stream>>>(esc, ei, emaxk, den, E,
                                                     ESL);
  edge_agg_kernel<<<edgeWaveBlk, blk, 0, stream>>>(esc, den, xl, ei, h2, E,
                                                   ESL);
  bias_relu_kernel<<<nodeElemBlk, blk, 0, stream>>>(h2, b2, N * EMB);

  // ---------------- Path readout ----------------
  zero_kernel<<<dim3(64), blk, 0, stream>>>((unsigned*)pemb, (size_t)128 * EMB);
  path_accum_kernel<<<dim3(512), blk, 0, stream>>>(h2, ei, masks, pemb, E);
  final_gemm_kernel<<<dim3(1), blk, 0, stream>>>(pemb, Wro, bro,
                                                 (float*)d_out);
}
